// ApertureAwareAttention_77343771066867
// MI455X (gfx1250) — matrix operand order repaired and checked by reading, not compiled, not run on hardware
//
#include <hip/hip_runtime.h>
#include <hip/hip_bf16.h>

// ---------------------------------------------------------------------------
// Axial attention (CSWin-like) for MI455X / gfx1250, wave32 + WMMA f16.
// B=4, H=W=128, C=192, NH=6, KD=32.  KD==32 matches v_wmma_f32_16x16x32_f16 K.
// ---------------------------------------------------------------------------

typedef __attribute__((ext_vector_type(16))) _Float16 v16h;
typedef __attribute__((ext_vector_type(8)))  _Float16 v8h;
typedef __attribute__((ext_vector_type(8)))  float    v8f;

#define BB 4
#define HH 128
#define WW 128
#define CC 192
#define NH 6
#define KD 32
#define MTOT (BB*HH*WW)          // 65536 rows for the projections
#define SCALING 0.17677669529663689f   // 32^-0.5

__device__ inline v8f zero8() {
    v8f z;
#pragma unroll
    for (int i = 0; i < 8; ++i) z[i] = 0.0f;
    return z;
}

// A fragment: 16x32 f16, row-major LDS, row stride `ld` halves.
// Per ISA 16-bit A layout: lane m = lane&15, half-select hs = lane>>4;
// a[0..7] = K = 8*hs + (0..7), a[8..15] = K = 16 + 8*hs + (0..7).
__device__ inline v16h frag_a_ld(const _Float16* base, int ld, int lane) {
    int m = lane & 15, hs = lane >> 4;
    const _Float16* p = base + m * ld + 8 * hs;
    v16h a;
#pragma unroll
    for (int t = 0; t < 8; ++t) { a[t] = p[t]; a[8 + t] = p[16 + t]; }
    return a;
}

// B fragment: 32x16 f16 with operand staged TRANSPOSED in LDS:
// baseT row n holds B[:,n] contiguous in K (stride ld halves).
// Lane n = lane&15: b[0..7] = K 8*hs+(0..7), b[8..15] = K 16+8*hs+(0..7).
__device__ inline v16h frag_bT_ld(const _Float16* baseT, int ld, int lane) {
    int n = lane & 15, hs = lane >> 4;
    const _Float16* p = baseT + n * ld + 8 * hs;
    v16h b;
#pragma unroll
    for (int t = 0; t < 8; ++t) { b[t] = p[t]; b[8 + t] = p[16 + t]; }
    return b;
}

// ---------------------------------------------------------------------------
// Projection GEMM: Y_f16[M,192] = (X_f32[M,192] @ W[192,192] + bias) * scale
// Block tile 128x64, 8 waves, K-loop 6x32.  All staging in 128-bit loads.
// ---------------------------------------------------------------------------
__global__ __launch_bounds__(256) void gemm_proj(
    const float* __restrict__ X, const float* __restrict__ Wm,
    const float* __restrict__ bias, float scale, _Float16* __restrict__ Yh)
{
    __shared__ __align__(16) _Float16 Xs[128 * 40];
    __shared__ __align__(16) _Float16 WsT[64 * 40];
    const int tid = threadIdx.x, lane = tid & 31, w0 = tid >> 5;
    const int mblk = blockIdx.x * 128, nblk = blockIdx.y * 64;

    v8f acc[4];
#pragma unroll
    for (int i = 0; i < 4; ++i) acc[i] = zero8();

    for (int kc = 0; kc < 192; kc += 32) {
        __syncthreads();
        // X tile 128x32: float4 loads, 4 per thread.
#pragma unroll
        for (int idx = tid; idx < 128 * 8; idx += 256) {
            int m = idx >> 3, p = idx & 7;
            float4 xv = *(const float4*)&X[(size_t)(mblk + m) * CC + kc + p * 4];
            _Float16* d = &Xs[m * 40 + p * 4];
            d[0] = (_Float16)xv.x; d[1] = (_Float16)xv.y;
            d[2] = (_Float16)xv.z; d[3] = (_Float16)xv.w;
            if (kc + 32 < 192)   // prefetch next K-chunk of this row
                __builtin_prefetch(&X[(size_t)(mblk + m) * CC + kc + 32 + p * 4], 0, 1);
        }
        // W tile 32x64 -> transposed LDS: float4 loads, 2 per thread.
#pragma unroll
        for (int idx = tid; idx < 32 * 16; idx += 256) {
            int k = idx >> 4, p = idx & 15;
            float4 wv = *(const float4*)&Wm[(size_t)(kc + k) * CC + nblk + p * 4];
            WsT[(p * 4 + 0) * 40 + k] = (_Float16)wv.x;
            WsT[(p * 4 + 1) * 40 + k] = (_Float16)wv.y;
            WsT[(p * 4 + 2) * 40 + k] = (_Float16)wv.z;
            WsT[(p * 4 + 3) * 40 + k] = (_Float16)wv.w;
        }
        __syncthreads();
        v16h a = frag_a_ld(Xs + (w0 * 16) * 40, 40, lane);
#pragma unroll
        for (int nt = 0; nt < 4; ++nt) {
            v16h b = frag_bT_ld(WsT + (nt * 16) * 40, 40, lane);
            acc[nt] = __builtin_amdgcn_wmma_f32_16x16x32_f16(
                false, a, false, b, (short)0, acc[nt], false, false);
        }
    }
    const int hs = lane >> 4, n = lane & 15;
#pragma unroll
    for (int nt = 0; nt < 4; ++nt) {
        int col = nblk + nt * 16 + n;
        float bb = bias[col];
#pragma unroll
        for (int r = 0; r < 8; ++r) {
            int row = mblk + w0 * 16 + hs * 8 + r;
            Yh[(size_t)row * CC + col] = (_Float16)((acc[nt][r] + bb) * scale);
        }
    }
}

// ---------------------------------------------------------------------------
// Output GEMM: Y_f32[M,192] = (X1_f16 + X2_f16)[M,192] @ Wo + bo
// ---------------------------------------------------------------------------
__global__ __launch_bounds__(256) void gemm_out(
    const _Float16* __restrict__ X1, const _Float16* __restrict__ X2,
    const float* __restrict__ Wm, const float* __restrict__ bias,
    float* __restrict__ Y)
{
    __shared__ __align__(16) _Float16 Xs[128 * 40];
    __shared__ __align__(16) _Float16 WsT[64 * 40];
    const int tid = threadIdx.x, lane = tid & 31, w0 = tid >> 5;
    const int mblk = blockIdx.x * 128, nblk = blockIdx.y * 64;

    v8f acc[4];
#pragma unroll
    for (int i = 0; i < 4; ++i) acc[i] = zero8();

    for (int kc = 0; kc < 192; kc += 32) {
        __syncthreads();
        // X tile: two 16-byte f16 loads + f32 add, 2 iters per thread.
#pragma unroll
        for (int idx = tid; idx < 128 * 4; idx += 256) {
            int m = idx >> 2, p = idx & 3;
            size_t g = (size_t)(mblk + m) * CC + kc + p * 8;
            v8h a = *(const v8h*)&X1[g];
            v8h b = *(const v8h*)&X2[g];
            v8h s;
#pragma unroll
            for (int j = 0; j < 8; ++j) s[j] = (_Float16)((float)a[j] + (float)b[j]);
            *(v8h*)&Xs[m * 40 + p * 8] = s;
            if (kc + 32 < 192)
                __builtin_prefetch(&X1[(size_t)(mblk + m) * CC + kc + 32 + p * 8], 0, 1);
        }
#pragma unroll
        for (int idx = tid; idx < 32 * 16; idx += 256) {
            int k = idx >> 4, p = idx & 15;
            float4 wv = *(const float4*)&Wm[(size_t)(kc + k) * CC + nblk + p * 4];
            WsT[(p * 4 + 0) * 40 + k] = (_Float16)wv.x;
            WsT[(p * 4 + 1) * 40 + k] = (_Float16)wv.y;
            WsT[(p * 4 + 2) * 40 + k] = (_Float16)wv.z;
            WsT[(p * 4 + 3) * 40 + k] = (_Float16)wv.w;
        }
        __syncthreads();
        v16h a = frag_a_ld(Xs + (w0 * 16) * 40, 40, lane);
#pragma unroll
        for (int nt = 0; nt < 4; ++nt) {
            v16h b = frag_bT_ld(WsT + (nt * 16) * 40, 40, lane);
            acc[nt] = __builtin_amdgcn_wmma_f32_16x16x32_f16(
                false, a, false, b, (short)0, acc[nt], false, false);
        }
    }
    const int hs = lane >> 4, n = lane & 15;
#pragma unroll
    for (int nt = 0; nt < 4; ++nt) {
        int col = nblk + nt * 16 + n;
        float bb = bias[col];
#pragma unroll
        for (int r = 0; r < 8; ++r) {
            int row = mblk + w0 * 16 + hs * 8 + r;
            Y[(size_t)row * CC + col] = acc[nt][r] + bb;
        }
    }
}

// ---------------------------------------------------------------------------
// Depthwise 5x5 conv (lepe) on v (f16 in, f16 out), pad 2.
// ---------------------------------------------------------------------------
__global__ __launch_bounds__(256) void lepe_conv(
    const _Float16* __restrict__ V, const float* __restrict__ kw,
    const float* __restrict__ kb, _Float16* __restrict__ out)
{
    int idx = blockIdx.x * 256 + threadIdx.x;     // over B*H*W*C exactly
    int c = idx % CC; int t = idx / CC;
    int w = t % WW; t /= WW;
    int h = t % HH; int b = t / HH;
    float s = kb[c];
#pragma unroll
    for (int ky = 0; ky < 5; ++ky) {
        int y = h + ky - 2;
        if (y < 0 || y >= HH) continue;
#pragma unroll
        for (int kx = 0; kx < 5; ++kx) {
            int x = w + kx - 2;
            if (x < 0 || x >= WW) continue;
            s += (float)V[(((size_t)b * HH + y) * WW + x) * CC + c] *
                 kw[(ky * 5 + kx) * CC + c];
        }
    }
    out[idx] = (_Float16)s;
}

// ---------------------------------------------------------------------------
// Axial attention, one block per (b, fixed, head); 256 threads = 8 waves.
// mode 0: row attention (fixed = h, sequence = w), V read like q/k, out = v1.
// mode 1: col attention (fixed = w, sequence = h), V = v1, out = attn result.
// Scores: 8 waves x 8 tiles of 16x16, one wmma each (K = KD = 32).
// Softmax: f32 in registers, width-16 shfl_xor butterflies (wave32 halves).
// PV: 2 N-tiles x 4 K-chunk wmmas.
// LDS total = 64512 B.
// ---------------------------------------------------------------------------
__global__ __launch_bounds__(256) void axial_attn(
    const _Float16* __restrict__ Q, const _Float16* __restrict__ K,
    const _Float16* __restrict__ V, const float* __restrict__ mask,
    _Float16* __restrict__ Out, int mode)
{
    __shared__ __align__(16) _Float16 qs[128 * 40];
    __shared__ __align__(16) _Float16 ks[128 * 40];
    __shared__ __align__(16) _Float16 vT[32 * 136];   // transposed V: [kd][seq]
    __shared__ __align__(16) _Float16 ps[128 * 136];  // exp probs
    __shared__ float rinv[128];

    const int tid = threadIdx.x, lane = tid & 31, w0 = tid >> 5;
    const int bid = blockIdx.x;
    const int head = bid % NH;
    int t = bid / NH;
    const int fixed = t % 128;
    const int b = t / 128;

    size_t qoff, voff, ooff;
    int qstr, vstr, ostr;
    const float* mk;
    if (mode == 0) {                       // row attention: fixed = h, seq = w
        qoff = ((size_t)(b * HH + fixed) * WW) * CC + head * KD;  qstr = CC;
        voff = qoff;                                              vstr = CC;
        mk = mask + head * WW * WW;
        // v1 layout: [b][w][head][h][kd]
        ooff = ((size_t)b * WW * NH + head) * (HH * KD) + (size_t)fixed * KD;
        ostr = NH * HH * KD;
    } else {                               // col attention: fixed = w, seq = h
        qoff = ((size_t)b * HH * WW + fixed) * CC + head * KD;    qstr = WW * CC;
        voff = ((size_t)(b * WW + fixed) * NH + head) * (HH * KD); vstr = KD;
        mk = mask + head * HH * HH;
        ooff = (size_t)b * HH * WW * CC + (size_t)fixed * CC + head * KD;
        ostr = WW * CC;
    }

    // Stage q, k row-major and V transposed into LDS (128-bit global loads).
#pragma unroll
    for (int idx = tid; idx < 128 * 4; idx += 256) {
        int s = idx >> 2, p = idx & 3;
        size_t gq = qoff + (size_t)s * qstr + p * 8;
        v8h qv = *(const v8h*)&Q[gq];
        v8h kv = *(const v8h*)&K[gq];
        *(v8h*)&qs[s * 40 + p * 8] = qv;
        *(v8h*)&ks[s * 40 + p * 8] = kv;
        v8h vv = *(const v8h*)&V[voff + (size_t)s * vstr + p * 8];
#pragma unroll
        for (int j = 0; j < 8; ++j) vT[(p * 8 + j) * 136 + s] = vv[j];
    }
    __syncthreads();

    const int qm = w0 * 16;                // this wave's query row block
    const int hs = lane >> 4, n = lane & 15;

    // ---- scores: S = q . k^T, one wmma per 16x16 tile ----
    v16h a = frag_a_ld(qs + qm * 40, 40, lane);
    v8f sc[8];
#pragma unroll
    for (int nt = 0; nt < 8; ++nt) {
        v16h bf = frag_bT_ld(ks + nt * 16 * 40, 40, lane);
        sc[nt] = __builtin_amdgcn_wmma_f32_16x16x32_f16(
            false, a, false, bf, (short)0, zero8(), false, false);
    }

    // ---- +mask, softmax (unnormalized exp to LDS, 1/sum kept) ----
#pragma unroll
    for (int r = 0; r < 8; ++r) {
        const int row = qm + hs * 8 + r;
        float m = -1e30f;
#pragma unroll
        for (int nt = 0; nt < 8; ++nt) {
            float s = sc[nt][r] + mk[row * 128 + nt * 16 + n];
            sc[nt][r] = s;
            m = fmaxf(m, s);
        }
#pragma unroll
        for (int d = 1; d < 16; d <<= 1) m = fmaxf(m, __shfl_xor(m, d, 32));
        float sum = 0.0f;
#pragma unroll
        for (int nt = 0; nt < 8; ++nt) {
            float p = __expf(sc[nt][r] - m);
            ps[row * 136 + nt * 16 + n] = (_Float16)p;
            sum += p;
        }
#pragma unroll
        for (int d = 1; d < 16; d <<= 1) sum += __shfl_xor(sum, d, 32);
        if (n == 0) rinv[row] = 1.0f / sum;
    }
    __syncthreads();

    // ---- PV: v1 = P . V  (2 N-tiles of kd, 4 K-chunks of 32 keys) ----
#pragma unroll
    for (int nt = 0; nt < 2; ++nt) {
        v8f acc = zero8();
#pragma unroll
        for (int kc = 0; kc < 4; ++kc) {
            v16h ap = frag_a_ld(ps + qm * 136 + kc * 32, 136, lane);
            v16h bp = frag_bT_ld(vT + nt * 16 * 136 + kc * 32, 136, lane);
            acc = __builtin_amdgcn_wmma_f32_16x16x32_f16(
                false, ap, false, bp, (short)0, acc, false, false);
        }
#pragma unroll
        for (int r = 0; r < 8; ++r) {
            const int row = qm + hs * 8 + r;
            Out[ooff + (size_t)row * ostr + nt * 16 + n] =
                (_Float16)(acc[r] * rinv[row]);
        }
    }
}

// ---------------------------------------------------------------------------
extern "C" void kernel_launch(void* const* d_in, const int* in_sizes, int n_in,
                              void* d_out, int out_size, void* d_ws, size_t ws_size,
                              hipStream_t stream) {
    const float* x      = (const float*)d_in[0];
    const float* mask_h = (const float*)d_in[1];
    const float* mask_w = (const float*)d_in[2];
    const float* Wq     = (const float*)d_in[3];
    const float* bq     = (const float*)d_in[4];
    const float* Wk     = (const float*)d_in[5];
    const float* bk     = (const float*)d_in[6];
    const float* Wv     = (const float*)d_in[7];
    const float* bv     = (const float*)d_in[8];
    const float* lepe_w = (const float*)d_in[9];
    const float* lepe_b = (const float*)d_in[10];
    const float* Wo     = (const float*)d_in[11];
    const float* bo     = (const float*)d_in[12];
    float* out = (float*)d_out;

    const size_t one = (size_t)MTOT * CC * sizeof(_Float16);   // 25,165,824 B
    char* p = (char*)d_ws;
    _Float16* qh    = (_Float16*)(p + 0 * one);
    _Float16* kh    = (_Float16*)(p + 1 * one);
    _Float16* vh    = (_Float16*)(p + 2 * one);
    _Float16* lepeh = (_Float16*)(p + 3 * one);
    _Float16* v1h   = (_Float16*)(p + 4 * one);  // [b][w][head][h][kd]
    _Float16* atth  = (_Float16*)(p + 5 * one);

    dim3 gg(MTOT / 128, CC / 64);                // (512, 3)
    gemm_proj<<<gg, 256, 0, stream>>>(x, Wq, bq, 1.0f,    qh);
    gemm_proj<<<gg, 256, 0, stream>>>(x, Wk, bk, SCALING, kh);
    gemm_proj<<<gg, 256, 0, stream>>>(x, Wv, bv, 1.0f,    vh);

    lepe_conv<<<(MTOT * CC) / 256, 256, 0, stream>>>(vh, lepe_w, lepe_b, lepeh);

    const int nblk = BB * 128 * NH;              // 3072
    axial_attn<<<nblk, 256, 0, stream>>>(qh, kh, vh,  mask_w, v1h,  0);
    axial_attn<<<nblk, 256, 0, stream>>>(qh, kh, v1h, mask_h, atth, 1);

    gemm_out<<<gg, 256, 0, stream>>>(atth, lepeh, Wo, bo, out);
}
